// Transformer_755914244508
// MI455X (gfx1250) — matrix operand order repaired and checked by reading, not compiled, not run on hardware
//
#include <hip/hip_runtime.h>

// ---------------------------------------------------------------------------
// CDNA5 (gfx1250) transformer block: LN -> MHA (flash) -> LN -> MLP(GELU)
// bf16 WMMA everywhere; async global->LDS tile staging with double buffering.
// ---------------------------------------------------------------------------

typedef __attribute__((ext_vector_type(16))) __bf16 bf16x16;
typedef __attribute__((ext_vector_type(8)))  __bf16 bf16x8;
typedef __attribute__((ext_vector_type(8)))  float  v8f;
typedef __attribute__((ext_vector_type(4)))  int    v4i;

#ifndef __has_builtin
#define __has_builtin(x) 0
#endif
#if __has_builtin(__builtin_amdgcn_global_load_async_to_lds_b128)
#define HAVE_ASYNC_LDS 1
#else
#define HAVE_ASYNC_LDS 0
#endif

__device__ __forceinline__ void async_b128(const void* g, void* l) {
#if HAVE_ASYNC_LDS
  __builtin_amdgcn_global_load_async_to_lds_b128(
      (__attribute__((address_space(1))) v4i*)g,
      (__attribute__((address_space(3))) v4i*)l, 0, 0);
#else
  *reinterpret_cast<uint4*>(l) = *reinterpret_cast<const uint4*>(g);
#endif
}

__device__ __forceinline__ void wait_async() {
#if HAVE_ASYNC_LDS
  asm volatile("s_wait_asynccnt 0x0" ::: "memory");
#endif
}

__device__ __forceinline__ v8f wmma_bf16(bf16x16 a, bf16x16 b, v8f c) {
  // 8 args: (neg_a, A, neg_b, B, c_mod, C, reuse_a, reuse_b)
  return __builtin_amdgcn_wmma_f32_16x16x32_bf16(false, a, false, b,
                                                 (short)0, c, false, false);
}

__device__ __forceinline__ bf16x16 cat8(bf16x8 lo, bf16x8 hi) {
  return __builtin_shufflevector(lo, hi, 0, 1, 2, 3, 4, 5, 6, 7,
                                 8, 9, 10, 11, 12, 13, 14, 15);
}

__device__ __forceinline__ v8f v8zero() {
  v8f z = {0.f, 0.f, 0.f, 0.f, 0.f, 0.f, 0.f, 0.f};
  return z;
}

__device__ __forceinline__ float gelu_exact(float x) {
  return 0.5f * x * (1.0f + erff(x * 0.70710678118654752f));
}

// ---------------------------------------------------------------------------
// Transpose + convert: W[K][N] (f32) -> Wt[N][K] (bf16). K%32==0, N%32==0.
// ---------------------------------------------------------------------------
__global__ __launch_bounds__(256) void transpose_bf16_k(
    const float* __restrict__ W, __bf16* __restrict__ Wt, int K, int N)
{
  __shared__ float tile[32][33];
  const int bn = blockIdx.x * 32;        // N offset
  const int bk = blockIdx.y * 32;        // K offset
  const int tx = threadIdx.x & 31;
  const int ty = threadIdx.x >> 5;       // 0..7
#pragma unroll
  for (int p = 0; p < 4; ++p)
    tile[ty + p * 8][tx] = W[(size_t)(bk + ty + p * 8) * N + bn + tx];
  __syncthreads();
#pragma unroll
  for (int p = 0; p < 4; ++p)
    Wt[(size_t)(bn + ty + p * 8) * K + bk + tx] =
        (__bf16)tile[tx][ty + p * 8];
}

// ---------------------------------------------------------------------------
// LayerNorm: one block (256 threads) per row of length D; bf16 output.
// ---------------------------------------------------------------------------
__global__ __launch_bounds__(256) void layernorm_k(
    const float* __restrict__ x, const float* __restrict__ g,
    const float* __restrict__ b, __bf16* __restrict__ out, int D)
{
  const int row = blockIdx.x;
  const int tid = threadIdx.x;
  const float* xr = x + (size_t)row * D;
  __bf16* orow = out + (size_t)row * D;
  __shared__ float red[256];

  float s = 0.f;
  for (int i = tid; i < D; i += 256) s += xr[i];
  red[tid] = s; __syncthreads();
  for (int st = 128; st > 0; st >>= 1) {
    if (tid < st) red[tid] += red[tid + st];
    __syncthreads();
  }
  const float mu = red[0] / (float)D;
  __syncthreads();

  float v = 0.f;
  for (int i = tid; i < D; i += 256) { float d = xr[i] - mu; v += d * d; }
  red[tid] = v; __syncthreads();
  for (int st = 128; st > 0; st >>= 1) {
    if (tid < st) red[tid] += red[tid + st];
    __syncthreads();
  }
  const float rstd = rsqrtf(red[0] / (float)D + 1e-5f);
  __syncthreads();

  for (int i = tid; i < D; i += 256)
    orow[i] = (__bf16)((xr[i] - mu) * rstd * g[i] + b[i]);
}

// ---------------------------------------------------------------------------
// GEMM: out = epilogue(A[M,K] @ Bt[N,K]^T + bias [, gelu] [, +res])
// A, Bt bf16 row-major; optional f32 and/or bf16 outputs.
// Block tile 128x128, K-step 32, double-buffered async LDS staging.
// 8 waves (2x4), each wave 64x32 = 4x2 WMMA tiles.
// ---------------------------------------------------------------------------
__global__ __launch_bounds__(256) void gemm_wmma(
    const __bf16* __restrict__ A, const __bf16* __restrict__ Bt,
    const float* __restrict__ bias, const float* __restrict__ res,
    float* __restrict__ Cf, __bf16* __restrict__ Cb,
    int M, int N, int K, int act)
{
  __shared__ __attribute__((aligned(16))) __bf16 lA[2][128][32];  // [m][k]
  __shared__ __attribute__((aligned(16))) __bf16 lB[2][128][32];  // [n][k]

  const int tid  = threadIdx.x;
  const int lane = tid & 31;
  const int wave = tid >> 5;
  const int hi   = lane >> 4;
  const int l16  = lane & 15;
  const int wm   = (wave >> 2) * 64;
  const int wn   = (wave & 3) * 32;
  const int bm   = blockIdx.y * 128;
  const int bn   = blockIdx.x * 128;

  const __bf16* Arow0 = A  + (size_t)bm * K;
  const __bf16* Brow0 = Bt + (size_t)bn * K;

  v8f acc[4][2];
#pragma unroll
  for (int i = 0; i < 4; ++i)
#pragma unroll
    for (int j = 0; j < 2; ++j) acc[i][j] = v8zero();

  auto stage = [&](int buf, int k0) {
#pragma unroll
    for (int t = 0; t < 2; ++t) {
      const int i = tid + t * 256;        // 0..511
      const int row = i >> 2, seg = (i & 3) * 8;
      async_b128(Arow0 + (size_t)row * K + k0 + seg, &lA[buf][row][seg]);
      async_b128(Brow0 + (size_t)row * K + k0 + seg, &lB[buf][row][seg]);
    }
  };

  const int nk = K / 32;
  stage(0, 0);
  wait_async();
  __syncthreads();

  for (int s = 0; s < nk; ++s) {
    const int cur = s & 1;
    if (s + 1 < nk) stage(cur ^ 1, (s + 1) * 32);

    bf16x16 afr[4], bfr[2];
#pragma unroll
    for (int i = 0; i < 4; ++i) {
      const __bf16* ap = &lA[cur][wm + i * 16 + l16][0];
      afr[i] = cat8(*(const bf16x8*)(ap + 8 * hi),
                    *(const bf16x8*)(ap + 16 + 8 * hi));
    }
#pragma unroll
    for (int j = 0; j < 2; ++j) {
      const __bf16* bp = &lB[cur][wn + j * 16 + l16][0];
      bfr[j] = cat8(*(const bf16x8*)(bp + 8 * hi),
                    *(const bf16x8*)(bp + 16 + 8 * hi));
    }
#pragma unroll
    for (int i = 0; i < 4; ++i)
#pragma unroll
      for (int j = 0; j < 2; ++j)
        acc[i][j] = wmma_bf16(afr[i], bfr[j], acc[i][j]);

    wait_async();
    __syncthreads();
  }

  // epilogue: bias (+gelu) (+residual), optional f32 / bf16 stores
#pragma unroll
  for (int i = 0; i < 4; ++i) {
#pragma unroll
    for (int j = 0; j < 2; ++j) {
      const int col  = bn + wn + j * 16 + l16;
      const int row0 = bm + wm + i * 16 + 8 * hi;
      const float bv = bias[col];
#pragma unroll
      for (int r = 0; r < 8; ++r) {
        float val = acc[i][j][r] + bv;
        if (act) val = gelu_exact(val);
        const size_t idx = (size_t)(row0 + r) * N + col;
        if (res) val += res[idx];
        if (Cf) Cf[idx] = val;
        if (Cb) Cb[idx] = (__bf16)val;
      }
    }
  }
}

// ---------------------------------------------------------------------------
// Flash attention: one wave per (batch, head, 16-query tile), key tiles of 32.
// Q/K/V bf16 in, ctx bf16 out (ctx may alias Q). Online softmax in f32.
// ---------------------------------------------------------------------------
__global__ __launch_bounds__(32) void attn_wmma(
    const __bf16* __restrict__ Q, const __bf16* __restrict__ Kc,
    const __bf16* __restrict__ Vc, __bf16* __restrict__ Ctx)
{
  const int T = 2048, H = 16, HD = 128, D = H * HD;
  const int q0 = blockIdx.x * 16;
  const int h  = blockIdx.y;
  const int b  = blockIdx.z;
  const int lane = threadIdx.x;
  const int hi = lane >> 4, l16 = lane & 15;

  __shared__ __attribute__((aligned(16))) __bf16 Pl[16 * 32];
  __shared__ __attribute__((aligned(16))) __bf16 Vl[32 * 128];

  // Q fragments: 4 chunks of K=32 over head_dim=128 (A-matrix layout).
  bf16x16 qa[4];
  {
    const __bf16* qrow = Q + (size_t)(b * T + q0 + l16) * D + h * HD;
#pragma unroll
    for (int c = 0; c < 4; ++c)
      qa[c] = cat8(*(const bf16x8*)(qrow + 32 * c + 8 * hi),
                   *(const bf16x8*)(qrow + 32 * c + 16 + 8 * hi));
  }

  v8f O[8];
#pragma unroll
  for (int j = 0; j < 8; ++j) O[j] = v8zero();
  float mi[8], li[8];
#pragma unroll
  for (int r = 0; r < 8; ++r) { mi[r] = -1e30f; li[r] = 0.f; }
  const float scl = 0.0883883476483184f;  // 1/sqrt(128)

  for (int s0 = 0; s0 <= q0 + 15; s0 += 32) {
    // ---- scores: Q(16x128) @ K^T(128x32), two 16-wide n-tiles ----
    v8f sc[2]; sc[0] = v8zero(); sc[1] = v8zero();
#pragma unroll
    for (int t = 0; t < 2; ++t) {
      const int ki = s0 + 16 * t + l16;
      const __bf16* krow = Kc + (size_t)(b * T + (ki < T ? ki : T - 1)) * HD;
#pragma unroll
      for (int c = 0; c < 4; ++c) {
        const bf16x16 kb = cat8(*(const bf16x8*)(krow + 32 * c + 8 * hi),
                                *(const bf16x8*)(krow + 32 * c + 16 + 8 * hi));
        sc[t] = wmma_bf16(qa[c], kb, sc[t]);
      }
    }

    // ---- scale + causal mask (row m=r+8*hi, col n=l16+16*t) ----
    float sm[2][8];
#pragma unroll
    for (int t = 0; t < 2; ++t) {
      const int ki = s0 + 16 * t + l16;
#pragma unroll
      for (int r = 0; r < 8; ++r) {
        const int qi = q0 + r + 8 * hi;
        float s = sc[t][r] * scl;
        if (ki > qi || ki >= T) s = -1e30f;
        sm[t][r] = s;
      }
    }

    // ---- online softmax stats (rows live per element, 16-lane halves) ----
    float esc[8];
#pragma unroll
    for (int r = 0; r < 8; ++r) {
      float mx = fmaxf(sm[0][r], sm[1][r]);
#pragma unroll
      for (int o = 8; o >= 1; o >>= 1) mx = fmaxf(mx, __shfl_xor(mx, o, 16));
      const float nm = fmaxf(mi[r], mx);
      const float p0 = __expf(sm[0][r] - nm);
      const float p1 = __expf(sm[1][r] - nm);
      sm[0][r] = p0; sm[1][r] = p1;
      float rs = p0 + p1;
#pragma unroll
      for (int o = 8; o >= 1; o >>= 1) rs += __shfl_xor(rs, o, 16);
      const float e0 = __expf(mi[r] - nm);
      li[r] = li[r] * e0 + rs;
      mi[r] = nm;
      esc[r] = e0;
    }
#pragma unroll
    for (int j = 0; j < 8; ++j)
#pragma unroll
      for (int r = 0; r < 8; ++r) O[j][r] *= esc[r];

    // ---- P (16x32) to LDS for A-layout reload ----
#pragma unroll
    for (int t = 0; t < 2; ++t)
#pragma unroll
      for (int r = 0; r < 8; ++r)
        Pl[(r + 8 * hi) * 32 + 16 * t + l16] = (__bf16)sm[t][r];

    // ---- stage V key-tile (32x128 bf16, 8KB) via async copies ----
    {
      const __bf16* vbase = Vc + (size_t)(b * T) * HD;
#pragma unroll
      for (int it = 0; it < 16; ++it) {
        const int lin = (it * 32 + lane) * 8;
        const int vr = lin >> 7, vcol = lin & 127;
        int vi = s0 + vr; if (vi > T - 1) vi = T - 1;
        async_b128(vbase + (size_t)vi * HD + vcol, &Vl[vr * 128 + vcol]);
      }
    }
    wait_async();
    __syncthreads();

    // ---- O += P(16x32) @ V(32x128) ----
    const __bf16* pp = &Pl[l16 * 32];
    const bf16x16 pa = cat8(*(const bf16x8*)(pp + 16 * hi),
                            *(const bf16x8*)(pp + 16 * hi + 8));
#pragma unroll
    for (int j = 0; j < 8; ++j) {
      bf16x16 vfr;
#pragma unroll
      for (int e = 0; e < 16; ++e)
        vfr[e] = Vl[(16 * hi + e) * 128 + j * 16 + l16];
      O[j] = wmma_bf16(pa, vfr, O[j]);
    }
    __syncthreads();
  }

  // ---- finalize ----
#pragma unroll
  for (int r = 0; r < 8; ++r) li[r] = 1.0f / li[r];
#pragma unroll
  for (int j = 0; j < 8; ++j)
#pragma unroll
    for (int r = 0; r < 8; ++r)
      Ctx[(size_t)(b * T + q0 + r + 8 * hi) * D + h * HD + j * 16 + l16] =
          (__bf16)(O[j][r] * li[r]);
}

// ---------------------------------------------------------------------------
// Host-side orchestration.
// ---------------------------------------------------------------------------
extern "C" void kernel_launch(void* const* d_in, const int* in_sizes, int n_in,
                              void* d_out, int out_size, void* d_ws, size_t ws_size,
                              hipStream_t stream)
{
  (void)in_sizes; (void)n_in; (void)out_size; (void)ws_size;

  const float* x   = (const float*)d_in[0];
  // d_in[1] = start_pos (== 0), unused
  const float* wq  = (const float*)d_in[2];
  const float* bq  = (const float*)d_in[3];
  const float* wk  = (const float*)d_in[4];
  const float* bk  = (const float*)d_in[5];
  const float* wv  = (const float*)d_in[6];
  const float* bv  = (const float*)d_in[7];
  const float* wo  = (const float*)d_in[8];
  const float* bo  = (const float*)d_in[9];
  const float* g1  = (const float*)d_in[10];
  const float* be1 = (const float*)d_in[11];
  const float* g2  = (const float*)d_in[12];
  const float* be2 = (const float*)d_in[13];
  const float* w1  = (const float*)d_in[14];
  const float* bb1 = (const float*)d_in[15];
  const float* w2  = (const float*)d_in[16];
  const float* bb2 = (const float*)d_in[17];

  const int Bz = 2, T = 2048, D = 2048, H = 16, HD = 128;
  const int M = Bz * T;     // 4096

  // ---- workspace carve-up (256B aligned chunks) ----
  size_t off = 0;
  char* base = (char*)d_ws;
  auto carve = [&](size_t bytes) -> void* {
    void* p = base + off;
    off += (bytes + 255) & ~(size_t)255;
    return p;
  };
  __bf16* wqT = (__bf16*)carve((size_t)D * D * 2);       // [2048][2048]
  __bf16* wkT = (__bf16*)carve((size_t)HD * D * 2);      // [128][2048]
  __bf16* wvT = (__bf16*)carve((size_t)HD * D * 2);
  __bf16* woT = (__bf16*)carve((size_t)D * D * 2);
  __bf16* w1T = (__bf16*)carve((size_t)4 * D * D * 2);   // [8192][2048]
  __bf16* w2T = (__bf16*)carve((size_t)D * 4 * D * 2);   // [2048][8192]
  __bf16* hb  = (__bf16*)carve((size_t)M * D * 2);       // LN out (reused)
  __bf16* qb  = (__bf16*)carve((size_t)M * D * 2);       // q, then ctx in-place
  __bf16* kb  = (__bf16*)carve((size_t)M * HD * 2);
  __bf16* vb2 = (__bf16*)carve((size_t)M * HD * 2);
  float*  x2  = (float*) carve((size_t)M * D * 4);       // post-attn residual
  __bf16* gb  = (__bf16*)carve((size_t)M * 4 * D * 2);   // MLP hidden

  float* xout = (float*)d_out;                           // (B,T,D)
  float* kout = xout + (size_t)M * D;                    // k_cat (B,1,T,HD)
  float* vout = kout + (size_t)M * HD;                   // v_cat

  // 0) one-shot weight transpose+convert to bf16 [N][K]
  transpose_bf16_k<<<dim3(D / 32,     D / 32),     256, 0, stream>>>(wq, wqT, D, D);
  transpose_bf16_k<<<dim3(HD / 32,    D / 32),     256, 0, stream>>>(wk, wkT, D, HD);
  transpose_bf16_k<<<dim3(HD / 32,    D / 32),     256, 0, stream>>>(wv, wvT, D, HD);
  transpose_bf16_k<<<dim3(D / 32,     D / 32),     256, 0, stream>>>(wo, woT, D, D);
  transpose_bf16_k<<<dim3(4 * D / 32, D / 32),     256, 0, stream>>>(w1, w1T, D, 4 * D);
  transpose_bf16_k<<<dim3(D / 32,     4 * D / 32), 256, 0, stream>>>(w2, w2T, 4 * D, D);

  // 1) h = LN1(x)  (bf16)
  layernorm_k<<<M, 256, 0, stream>>>(x, g1, be1, hb, D);

  // 2) q/k/v projections; k,v also written f32 straight into d_out cache slots
  gemm_wmma<<<dim3(D / 128, M / 128), 256, 0, stream>>>(hb, wqT, bq, nullptr,
                                                        nullptr, qb, M, D, D, 0);
  gemm_wmma<<<dim3(1, M / 128), 256, 0, stream>>>(hb, wkT, bk, nullptr,
                                                  kout, kb, M, HD, D, 0);
  gemm_wmma<<<dim3(1, M / 128), 256, 0, stream>>>(hb, wvT, bv, nullptr,
                                                  vout, vb2, M, HD, D, 0);

  // 3) flash attention; ctx aliases q (per-block disjoint, reads-then-writes)
  attn_wmma<<<dim3(T / 16, H, Bz), 32, 0, stream>>>(qb, kb, vb2, qb);

  // 4) x2 = x + ctx@wo + bo
  gemm_wmma<<<dim3(D / 128, M / 128), 256, 0, stream>>>(qb, woT, bo, x,
                                                        x2, nullptr, M, D, D, 0);

  // 5) h2 = LN2(x2)
  layernorm_k<<<M, 256, 0, stream>>>(x2, g2, be2, hb, D);

  // 6) g = gelu(h2@w1 + b1)  (bf16)
  gemm_wmma<<<dim3(4 * D / 128, M / 128), 256, 0, stream>>>(hb, w1T, bb1, nullptr,
                                                            nullptr, gb, M, 4 * D, D, 1);

  // 7) x_out = x2 + g@w2 + b2
  gemm_wmma<<<dim3(D / 128, M / 128), 256, 0, stream>>>(gb, w2T, bb2, x2,
                                                        xout, nullptr, M, D, 4 * D, 0);
}
